// DropoutTransformer_80444737454622
// MI455X (gfx1250) — matrix operand order repaired and checked by reading, not compiled, not run on hardware
//
#include <hip/hip_runtime.h>
#include <math.h>

// ---------------- types for WMMA fragments ----------------
typedef __attribute__((ext_vector_type(16))) __bf16   bf16x16;
typedef __attribute__((ext_vector_type(8)))  float    f32x8;
typedef __attribute__((ext_vector_type(8)))  unsigned short us8;

union Frag16 {
    struct { us8 lo; us8 hi; } u;
    bf16x16 b;
};

__device__ __forceinline__ unsigned short f2bf(float f) {
    unsigned int u = __float_as_uint(f);
    u += 0x7FFFu + ((u >> 16) & 1u);      // round-to-nearest-even
    return (unsigned short)(u >> 16);
}

// ---------------- embedding: h = w_out[x] + pos_emb ----------------
__global__ __launch_bounds__(256) void embed_kernel(
    const int* __restrict__ x, const float* __restrict__ wout,
    const float* __restrict__ pos, float* __restrict__ h,
    int T, int C, long total)
{
    long i = (long)blockIdx.x * 256 + threadIdx.x;
    if (i >= total) return;
    long bt = i / C;
    int  c  = (int)(i - bt * C);
    int  t  = (int)(bt % T);
    int  tok = x[bt];
    h[i] = wout[(long)tok * C + c] + pos[(long)t * C + c];
}

// ---------------- weight convert / transpose to bf16 ----------------
// dst[c * R + r] = bf16(src[r * Ccol + c])   (i.e. store N x K row-major)
__global__ __launch_bounds__(256) void transpose_bf16_kernel(
    const float* __restrict__ src, unsigned short* __restrict__ dst,
    int R, int Ccol)
{
    long i = (long)blockIdx.x * 256 + threadIdx.x;
    long total = (long)R * Ccol;
    if (i >= total) return;
    long r = i / Ccol;
    long c = i - r * Ccol;
    dst[c * R + r] = f2bf(src[i]);
}

__global__ __launch_bounds__(256) void convert_bf16_kernel(
    const float* __restrict__ src, unsigned short* __restrict__ dst, long n)
{
    long i = (long)blockIdx.x * 256 + threadIdx.x;
    if (i < n) dst[i] = f2bf(src[i]);
}

// ---------------- per-row sum of squares (w_sq) ----------------
__global__ __launch_bounds__(256) void rowsq_kernel(
    const float* __restrict__ w, float* __restrict__ out, int C)
{
    __shared__ float red[256];
    int row = blockIdx.x, tid = threadIdx.x;
    float s = 0.f;
    for (int c = tid; c < C; c += 256) { float v = w[(long)row * C + c]; s += v * v; }
    red[tid] = s; __syncthreads();
    for (int off = 128; off > 0; off >>= 1) {
        if (tid < off) red[tid] += red[tid + off];
        __syncthreads();
    }
    if (tid == 0) out[row] = red[0];
}

// ---------------- layernorm: bf16 out, optional xsq ----------------
__global__ __launch_bounds__(256) void layernorm_kernel(
    const float* __restrict__ h, const float* __restrict__ g,
    unsigned short* __restrict__ outBF, float* __restrict__ xsq, int C)
{
    __shared__ float red[256];
    int row = blockIdx.x, tid = threadIdx.x;
    const float* hp = h + (long)row * C;

    float s = 0.f;
    for (int c = tid; c < C; c += 256) s += hp[c];
    red[tid] = s; __syncthreads();
    for (int off = 128; off > 0; off >>= 1) { if (tid < off) red[tid] += red[tid + off]; __syncthreads(); }
    float mu = red[0] / C; __syncthreads();

    float v = 0.f;
    for (int c = tid; c < C; c += 256) { float d = hp[c] - mu; v += d * d; }
    red[tid] = v; __syncthreads();
    for (int off = 128; off > 0; off >>= 1) { if (tid < off) red[tid] += red[tid + off]; __syncthreads(); }
    float rstd = rsqrtf(red[0] / C + 1e-5f); __syncthreads();

    float sq = 0.f;
    for (int c = tid; c < C; c += 256) {
        float val = (hp[c] - mu) * rstd * g[c];
        outBF[(long)row * C + c] = f2bf(val);
        sq += val * val;
    }
    if (xsq) {
        red[tid] = sq; __syncthreads();
        for (int off = 128; off > 0; off >>= 1) { if (tid < off) red[tid] += red[tid + off]; __syncthreads(); }
        if (tid == 0) xsq[row] = red[0];
    }
}

// ---------------- WMMA bf16 GEMM helpers ----------------
__device__ __forceinline__ void load_step(
    const unsigned short* __restrict__ A, long a_off,
    const unsigned short* __restrict__ Bt, const long* b_off,
    int k, int hi, Frag16& af, Frag16* bfr)
{
    af.u.lo = *(const us8*)(A + a_off + k + hi * 8);
    af.u.hi = *(const us8*)(A + a_off + k + 16 + hi * 8);
#pragma unroll
    for (int t = 0; t < 4; ++t) {
        const unsigned short* bp = Bt + b_off[t] + k + hi * 8;
        bfr[t].u.lo = *(const us8*)(bp);
        bfr[t].u.hi = *(const us8*)(bp + 16);
    }
}

__device__ __forceinline__ void mma_step(const Frag16& af, const Frag16* bfr, f32x8* acc)
{
#pragma unroll
    for (int t = 0; t < 4; ++t)
        acc[t] = __builtin_amdgcn_wmma_f32_16x16x32_bf16(
            false, af.b, false, bfr[t].b, (short)0, acc[t], false, false);
}

// ---------------- WMMA bf16 GEMM  C[M,N] = A[M,K] @ Bt[N,K]^T ----------------
// Requires K % 64 == 0 (true for K = 768 and 3072 here).
// mode 0: outF = acc (+ residual)            (fp32)
// mode 1: outBF = bf16(gelu_exact(acc))
// mode 3: outF = (wsq[n] - 2*acc + xsq[m]) / 768   (logits epilogue)
__global__ __launch_bounds__(256) void gemm_bf16_kernel(
    const unsigned short* __restrict__ A,
    const unsigned short* __restrict__ Bt,
    float* __restrict__ outF,
    unsigned short* __restrict__ outBF,
    const float* __restrict__ residual,
    const float* __restrict__ wsq,
    const float* __restrict__ xsq,
    int M, int N, int K, int mode)
{
    const int lane = threadIdx.x & 31;
    const int wave = threadIdx.x >> 5;
    const int wm   = wave >> 2;          // 0..1
    const int wn   = wave & 3;           // 0..3
    const int hi   = (lane >> 4) & 1;
    const int l16  = lane & 15;

    const int m_tile = blockIdx.y * 32 + wm * 16;
    const int n_base = blockIdx.x * 256 + wn * 64;
    if (m_tile >= M) return;             // wave-uniform: EXEC stays all-ones

    int a_row = m_tile + l16; if (a_row >= M) a_row = M - 1;
    const long a_off = (long)a_row * K;

    long b_off[4];
#pragma unroll
    for (int t = 0; t < 4; ++t) {
        int r = n_base + t * 16 + l16;
        if (r >= N) r = N - 1;           // clamp; garbage rows never written
        b_off[t] = (long)r * K;
    }

    f32x8 acc[4] = {};

    // Software-pipelined main loop, 64 of K per iteration, register double-buffer.
    // All loads for the next 32-chunk are issued before consuming the current one,
    // so the per-WMMA waits keep the prefetch loads outstanding (in-order LOADcnt).
    Frag16 af0, af1, bf0[4], bf1[4];
    load_step(A, a_off, Bt, b_off, 0, hi, af0, bf0);
    for (int k = 0; k < K; k += 64) {
        load_step(A, a_off, Bt, b_off, k + 32, hi, af1, bf1);
        mma_step(af0, bf0, acc);
        if (k + 64 < K)
            load_step(A, a_off, Bt, b_off, k + 64, hi, af0, bf0);
        mma_step(af1, bf1, acc);
    }

#pragma unroll
    for (int t = 0; t < 4; ++t) {
        int col = n_base + t * 16 + l16;
        if (col >= N) continue;
#pragma unroll
        for (int r = 0; r < 8; ++r) {
            int row = m_tile + hi * 8 + r;
            if (row >= M) continue;
            long idx = (long)row * N + col;
            float v = acc[t][r];
            if (mode == 0) {
                if (residual) v += residual[idx];
                outF[idx] = v;
            } else if (mode == 1) {
                float gl = 0.5f * v * (1.0f + erff(v * 0.70710678118654752f));
                outBF[idx] = f2bf(gl);
            } else { // mode 3: logits
                outF[idx] = (wsq[col] - 2.0f * v + xsq[row]) * (1.0f / 768.0f);
            }
        }
    }
}

// ---------------- causal attention (VALU), one wave per (b,h,q) ----------------
#define HD 64
__global__ __launch_bounds__(256) void attention_kernel(
    const float* __restrict__ qkv, unsigned short* __restrict__ oBF,
    int T, int H, int C3, int C, float scale)
{
    __shared__ float sc[8][1024];
    const int wave = threadIdx.x >> 5;
    const int lane = threadIdx.x & 31;
    long row_id = (long)blockIdx.x * 8 + wave;

    int  q  = (int)(row_id % T);
    long bh = row_id / T;
    int  h  = (int)(bh % H);
    int  b  = (int)(bh / H);

    const float* qp = qkv + ((long)b * T + q) * C3 + h * HD;
    float qreg[HD];
#pragma unroll
    for (int d = 0; d < HD; ++d) qreg[d] = qp[d];

    float m = -INFINITY;
    for (int j = lane; j <= q; j += 32) {
        const float* kp = qkv + ((long)b * T + j) * C3 + C + h * HD;
        float s = 0.f;
#pragma unroll
        for (int d = 0; d < HD; ++d) s += qreg[d] * kp[d];
        s *= scale;
        sc[wave][j] = s;
        m = fmaxf(m, s);
    }
    for (int off = 16; off > 0; off >>= 1) m = fmaxf(m, __shfl_xor(m, off, 32));

    float l = 0.f;
    for (int j = lane; j <= q; j += 32) {
        float p = expf(sc[wave][j] - m);
        sc[wave][j] = p;
        l += p;
    }
    for (int off = 16; off > 0; off >>= 1) l += __shfl_xor(l, off, 32);
    float inv = 1.0f / l;

    float a0 = 0.f, a1 = 0.f;
    for (int j = 0; j <= q; ++j) {
        float p = sc[wave][j];
        const float* vp = qkv + ((long)b * T + j) * C3 + 2 * C + h * HD;
        a0 += p * vp[lane];
        a1 += p * vp[lane + 32];
    }
    unsigned short* op = oBF + ((long)b * T + q) * C + h * HD;
    op[lane]      = f2bf(a0 * inv);
    op[lane + 32] = f2bf(a1 * inv);
}

// ---------------- host orchestration ----------------
extern "C" void kernel_launch(void* const* d_in, const int* in_sizes, int n_in,
                              void* d_out, int out_size, void* d_ws, size_t ws_size,
                              hipStream_t stream) {
    (void)in_sizes; (void)n_in; (void)out_size; (void)ws_size;
    const int*   x      = (const int*)  d_in[0];
    const float* w_out  = (const float*)d_in[1];
    const float* pos    = (const float*)d_in[2];
    const float* qkv_w  = (const float*)d_in[3];
    const float* proj_w = (const float*)d_in[4];
    const float* ln1_w  = (const float*)d_in[5];
    const float* ln2_w  = (const float*)d_in[6];
    const float* ff1_w  = (const float*)d_in[7];
    const float* ff2_w  = (const float*)d_in[8];
    const float* lnf_w  = (const float*)d_in[9];
    float* logits = (float*)d_out;

    const int B = 2, T = 1024, C = 768, V = 50257, L = 4, H = 12;
    const int BT = B * T, C3 = 3 * C, C4 = 4 * C;

    char* wsp = (char*)d_ws;
    auto alloc = [&](size_t bytes) -> char* {
        char* p = wsp;
        wsp += (bytes + 255) & ~(size_t)255;
        return p;
    };

    float*          h      = (float*)         alloc((size_t)BT * C  * 4);
    float*          qkv    = (float*)         alloc((size_t)BT * C3 * 4);
    unsigned short* a_bf   = (unsigned short*)alloc((size_t)BT * C  * 2);
    unsigned short* o_bf   = (unsigned short*)alloc((size_t)BT * C  * 2);
    unsigned short* g_bf   = (unsigned short*)alloc((size_t)BT * C4 * 2);
    unsigned short* hn_bf  = (unsigned short*)alloc((size_t)BT * C  * 2);
    float*          xsq    = (float*)         alloc((size_t)BT * 4);
    float*          wsq    = (float*)         alloc((size_t)V  * 4);
    unsigned short* woutBF = (unsigned short*)alloc((size_t)V  * C  * 2);
    unsigned short* qkvT   = (unsigned short*)alloc((size_t)L * C3 * C * 2);
    unsigned short* projT  = (unsigned short*)alloc((size_t)L * C  * C * 2);
    unsigned short* ff1T   = (unsigned short*)alloc((size_t)L * C4 * C * 2);
    unsigned short* ff2T   = (unsigned short*)alloc((size_t)L * C  * C4 * 2);

    auto cdiv = [](long a, long b) { return (int)((a + b - 1) / b); };

    // --- weight prep: bf16 (+transpose to NxK) ---
    convert_bf16_kernel<<<cdiv((long)V * C, 256), 256, 0, stream>>>(w_out, woutBF, (long)V * C);
    for (int l = 0; l < L; ++l) {
        transpose_bf16_kernel<<<cdiv((long)C * C3, 256), 256, 0, stream>>>(
            qkv_w + (long)l * C * C3, qkvT + (long)l * C3 * C, C, C3);
        transpose_bf16_kernel<<<cdiv((long)C * C, 256), 256, 0, stream>>>(
            proj_w + (long)l * C * C, projT + (long)l * C * C, C, C);
        transpose_bf16_kernel<<<cdiv((long)C * C4, 256), 256, 0, stream>>>(
            ff1_w + (long)l * C * C4, ff1T + (long)l * C4 * C, C, C4);
        transpose_bf16_kernel<<<cdiv((long)C4 * C, 256), 256, 0, stream>>>(
            ff2_w + (long)l * C4 * C, ff2T + (long)l * C * C4, C4, C);
    }
    rowsq_kernel<<<V, 256, 0, stream>>>(w_out, wsq, C);

    // --- embedding ---
    embed_kernel<<<cdiv((long)BT * C, 256), 256, 0, stream>>>(x, w_out, pos, h, T, C, (long)BT * C);

    const float scale = 0.125f; // hd^-0.5, hd=64
    dim3 blk(256);

    for (int l = 0; l < L; ++l) {
        // ln1 -> a_bf
        layernorm_kernel<<<BT, 256, 0, stream>>>(h, ln1_w + (long)l * C, a_bf, nullptr, C);
        // qkv = a @ qkv_w   (M=BT, N=3C, K=C)
        gemm_bf16_kernel<<<dim3(cdiv(C3, 256), cdiv(BT, 32)), blk, 0, stream>>>(
            a_bf, qkvT + (long)l * C3 * C, qkv, nullptr, nullptr, nullptr, nullptr,
            BT, C3, C, 0);
        // attention -> o_bf
        attention_kernel<<<(B * H * T) / 8, 256, 0, stream>>>(qkv, o_bf, T, H, C3, C, scale);
        // h = h + o @ proj_w   (in place)
        gemm_bf16_kernel<<<dim3(cdiv(C, 256), cdiv(BT, 32)), blk, 0, stream>>>(
            o_bf, projT + (long)l * C * C, h, nullptr, h, nullptr, nullptr,
            BT, C, C, 0);
        // ln2 -> a_bf
        layernorm_kernel<<<BT, 256, 0, stream>>>(h, ln2_w + (long)l * C, a_bf, nullptr, C);
        // g = gelu(a @ ff1) -> g_bf
        gemm_bf16_kernel<<<dim3(cdiv(C4, 256), cdiv(BT, 32)), blk, 0, stream>>>(
            a_bf, ff1T + (long)l * C4 * C, nullptr, g_bf, nullptr, nullptr, nullptr,
            BT, C4, C, 1);
        // h = h + g @ ff2   (in place, K=4C)
        gemm_bf16_kernel<<<dim3(cdiv(C, 256), cdiv(BT, 32)), blk, 0, stream>>>(
            g_bf, ff2T + (long)l * C * C4, h, nullptr, h, nullptr, nullptr,
            BT, C, C4, 0);
    }

    // final LN -> hn_bf + xsq
    layernorm_kernel<<<BT, 256, 0, stream>>>(h, lnf_w, hn_bf, xsq, C);

    // logits = (wsq - 2 * hn @ w_out^T + xsq) / C    (M=BT, N=V, K=C)
    gemm_bf16_kernel<<<dim3(cdiv(V, 256), cdiv(BT, 32)), blk, 0, stream>>>(
        hn_bf, woutBF, logits, nullptr, nullptr, wsq, xsq,
        BT, V, C, 3);
}
